// IntConv2d_87290915323933
// MI455X (gfx1250) — 8-bit matrix operand order repaired and checked by reading, compiled, not run on hardware
//
#include <hip/hip_runtime.h>
#include <hip/hip_bf16.h>

typedef __attribute__((ext_vector_type(8))) int v8i;
typedef __attribute__((ext_vector_type(4))) int v4i;
typedef __attribute__((ext_vector_type(2))) int v2i;

#define IN_C   64
#define OUT_C  128
#define KSZ    3
#define HH     56
#define WW     56
#define BATCH  64

#define LDS_COLS      66
#define LDS_BYTES     (3 * LDS_COLS * IN_C)

__device__ __forceinline__ unsigned pack4_perm(v4i v) {
    unsigned t0 = __builtin_amdgcn_perm((unsigned)v[1], (unsigned)v[0],
                                        0x04040400u);
    unsigned t1 = __builtin_amdgcn_perm((unsigned)v[3], (unsigned)v[2],
                                        0x04040400u);
    return __builtin_amdgcn_perm(t1, t0, 0x05040100u);
}

__global__ void pack_i8_kernel(const int* __restrict__ in,
                               unsigned* __restrict__ out, int n4) {
    int t = blockIdx.x * blockDim.x + threadIdx.x;
    if (t >= n4) return;
    out[t] = pack4_perm(((const v4i*)in)[t]);
}

__global__ __launch_bounds__(256)
void conv_i8_wmma_kernel(const int* __restrict__ x32,
                         const signed char* __restrict__ w8,
                         const int* __restrict__ bias32,
                         float* __restrict__ out) {
    __shared__ signed char lds[LDS_BYTES];

    const int tid  = threadIdx.x;
    const int wave = tid >> 5;
    const int lane = tid & 31;
    const int n    = lane & 15;
    const int h2   = lane >> 4;

    const int bh = blockIdx.x;
    const int b  = bh / HH;
    const int h  = bh - b * HH;

    for (int t = tid; t < 3 * LDS_COLS * 4; t += 256) {
        int r   = t / (LDS_COLS * 4);
        int rem = t - r * (LDS_COLS * 4);
        int col = rem >> 2;
        int cp  = rem & 3;
        int ih  = h - 1 + r;
        int iw  = col - 1;
        v4i val = {};
        if ((unsigned)ih < (unsigned)HH && (unsigned)iw < (unsigned)WW) {
            const v4i* gp = (const v4i*)(x32 +
                (((size_t)b * HH + ih) * WW + iw) * IN_C + cp * 16);
            v4i c0 = gp[0], c1 = gp[1], c2 = gp[2], c3 = gp[3];
            val[0] = (int)pack4_perm(c0);
            val[1] = (int)pack4_perm(c1);
            val[2] = (int)pack4_perm(c2);
            val[3] = (int)pack4_perm(c3);
        }
        *(v4i*)(&lds[(r * LDS_COLS + col) * IN_C + cp * 16]) = val;
    }
    __syncthreads();

    const int oc0 = wave * 16;
    v8i Bm[9];
    {
        const signed char* wb = w8 + (size_t)(oc0 + n) * (KSZ * KSZ * IN_C)
                                   + h2 * 16;
        #pragma unroll
        for (int t = 0; t < 9; ++t) {
            v4i lo = *(const v4i*)(wb + t * IN_C);
            v4i hi = *(const v4i*)(wb + t * IN_C + 32);
            Bm[t][0] = lo[0]; Bm[t][1] = lo[1]; Bm[t][2] = lo[2]; Bm[t][3] = lo[3];
            Bm[t][4] = hi[0]; Bm[t][5] = hi[1]; Bm[t][6] = hi[2]; Bm[t][7] = hi[3];
        }
    }
    const int bv = bias32[oc0 + n];

    for (int mt = 0; mt < 4; ++mt) {
        const int w0 = mt * 16;
        v8i acc = {};
        #pragma unroll
        for (int kh = 0; kh < 3; ++kh) {
            #pragma unroll
            for (int kw = 0; kw < 3; ++kw) {
                const signed char* ab =
                    &lds[(kh * LDS_COLS + (w0 + n + kw)) * IN_C + 16 * h2];
                v2i d0 = *(const v2i*)(ab +  0);
                v2i d1 = *(const v2i*)(ab +  8);
                v2i d2 = *(const v2i*)(ab + 32);
                v2i d3 = *(const v2i*)(ab + 40);
                v8i A;
                A[0] = d0[0]; A[1] = d0[1];
                A[2] = d1[0]; A[3] = d1[1];
                A[4] = d2[0]; A[5] = d2[1];
                A[6] = d3[0]; A[7] = d3[1];
                acc = __builtin_amdgcn_wmma_i32_16x16x64_iu8(
                     true, A,  true, Bm[kh * 3 + kw],
                    acc,  false,  false);
            }
        }
        float* ob = out + (((size_t)b * HH + h) * WW) * OUT_C + oc0 + n;
        #pragma unroll
        for (int r = 0; r < 8; ++r) {
            int wout = w0 + r + 8 * h2;
            if (wout < WW) {
                ob[(size_t)wout * OUT_C] = (float)(acc[r] + bv);
            }
        }
    }
}

extern "C" void kernel_launch(void* const* d_in, const int* in_sizes, int n_in,
                              void* d_out, int out_size, void* d_ws, size_t ws_size,
                              hipStream_t stream) {
    const int* x32 = (const int*)d_in[0];
    const int* w32 = (const int*)d_in[1];
    const int* b32 = (const int*)d_in[2];
    float* out = (float*)d_out;

    const int w_elems = OUT_C * KSZ * KSZ * IN_C;

    signed char* w8 = (signed char*)d_ws;

    {
        int n4 = w_elems / 4;
        pack_i8_kernel<<<(n4 + 255) / 256, 256, 0, stream>>>(
            w32, (unsigned*)w8, n4);
    }

    conv_i8_wmma_kernel<<<BATCH * HH, 256, 0, stream>>>(x32, w8, b32, out);
}
